// MultiHeadAttention_79405355369124
// MI455X (gfx1250) — compile-verified, matrix operand order checked by reading, not run on hardware
//
#include <hip/hip_runtime.h>
#include <hip/hip_bf16.h>

// ---------------------------------------------------------------------------
// MI455X (gfx1250) multi-head attention forward, f16 WMMA pipeline.
// 137 GFLOP vs ~6.5us of minimal HBM traffic at 23.3 TB/s -> tensor-core
// bound. All four matmul stages run on v_wmma_f32_16x16x32_f16.
// Round 5: softmax row reductions as fused VOP2+DPP16 (v_max_num_f32_dpp /
// v_add_f32_dpp: one instruction per butterfly step instead of mov+max).
// ---------------------------------------------------------------------------

typedef __attribute__((ext_vector_type(16))) _Float16 v16h;
typedef __attribute__((ext_vector_type(8)))  float    v8f;
typedef __attribute__((ext_vector_type(4)))  float    f32x4;   // trivial (union-safe)

static __device__ __forceinline__ v8f wmma16(v16h a, v16h b, v8f c) {
  // D = A(16x32 f16) * B(32x16 f16) + C(16x16 f32)
  return __builtin_amdgcn_wmma_f32_16x16x32_f16(false, a, false, b, (short)0, c,
                                                false, false);
}

// Butterfly reduction across each 16-lane half; all 16 lanes get the result.
// VOP2 DPP applies the lane-select to SRC0, so each step is a single
// instruction: dst = op(dpp(src), src). Syntax mirrors compiler disasm output.
static __device__ __forceinline__ float red16_max(float v) {
  float t;
  asm("v_max_num_f32_dpp %0, %1, %1 quad_perm:[1,0,3,2] row_mask:0xf bank_mask:0xf"
      : "=v"(t) : "v"(v));
  asm("v_max_num_f32_dpp %0, %1, %1 quad_perm:[2,3,0,1] row_mask:0xf bank_mask:0xf"
      : "=v"(v) : "v"(t));
  asm("v_max_num_f32_dpp %0, %1, %1 row_half_mirror row_mask:0xf bank_mask:0xf"
      : "=v"(t) : "v"(v));
  asm("v_max_num_f32_dpp %0, %1, %1 row_mirror row_mask:0xf bank_mask:0xf"
      : "=v"(v) : "v"(t));
  return v;
}
static __device__ __forceinline__ float red16_sum(float v) {
  float t;
  asm("v_add_f32_dpp %0, %1, %1 quad_perm:[1,0,3,2] row_mask:0xf bank_mask:0xf"
      : "=v"(t) : "v"(v));
  asm("v_add_f32_dpp %0, %1, %1 quad_perm:[2,3,0,1] row_mask:0xf bank_mask:0xf"
      : "=v"(v) : "v"(t));
  asm("v_add_f32_dpp %0, %1, %1 row_half_mirror row_mask:0xf bank_mask:0xf"
      : "=v"(t) : "v"(v));
  asm("v_add_f32_dpp %0, %1, %1 row_mirror row_mask:0xf bank_mask:0xf"
      : "=v"(v) : "v"(t));
  return v;
}

// Fragment gather shared by A-operands (row-major rows) and B-operands
// (columns, staged transposed in LDS so a "column" is a contiguous row).
// Per ISA 7.12.2 (16-bit 16x32 A layout, B symmetric):
//   lanes 0-15 : row passed in, k = {0..7} and {16..23}
//   lanes 16-31: row passed in, k = {8..15} and {24..31}
// Caller passes row = base_row + (lane & 15).
static __device__ __forceinline__ v16h load_frag(const _Float16* base,
                                                 int stride_h, int row, int lane) {
  const int hi = (lane >> 4) & 1;                     // k-half select
  const _Float16* p = base + row * stride_h + hi * 8;
  union { v16h v; struct { f32x4 a, b; } f; } u;
  u.f.a = *(const f32x4*)(p);        // k = hi*8 .. hi*8+7
  u.f.b = *(const f32x4*)(p + 16);   // k = 16+hi*8 .. 16+hi*8+7
  return u.v;
}

// ---------------------------------------------------------------------------
// f32 -> f16 conversion (8 floats in, 8 halfs out = one b128 store)
// ---------------------------------------------------------------------------
__global__ void cvt_f32_f16(const float* __restrict__ in,
                            _Float16* __restrict__ out, int n8) {
  int i = blockIdx.x * blockDim.x + threadIdx.x;
  if (i < n8) {
    f32x4 a = ((const f32x4*)in)[2 * i];
    f32x4 b = ((const f32x4*)in)[2 * i + 1];
    union { _Float16 h[8]; f32x4 v; } o;
    o.h[0] = (_Float16)a.x; o.h[1] = (_Float16)a.y;
    o.h[2] = (_Float16)a.z; o.h[3] = (_Float16)a.w;
    o.h[4] = (_Float16)b.x; o.h[5] = (_Float16)b.y;
    o.h[6] = (_Float16)b.z; o.h[7] = (_Float16)b.w;
    ((f32x4*)out)[i] = o.v;
  }
}

// ---------------------------------------------------------------------------
// WMMA GEMM: C(M x N) = A(M x K) * B(K x N) + bias
//   128x256 block tile, BK=64, 256 threads = 8 waves (2 x 4),
//   wave tile 64x64 = 4x4 WMMA accumulators (128 VGPRs/lane).
//   Register-pipelined LDS staging -> 32 WMMAs per barrier period per wave.
// MODE 0: QKV projection -> scatter to Q/K/V f16 (b,h,s,d),
//         Q pre-scaled by (1/sqrt(64)) * log2(e) for exp2-domain softmax.
// MODE 1: output projection -> f32 + bias to d_out.
// ---------------------------------------------------------------------------
template <int MODE>
__global__ __launch_bounds__(256) void gemm_f16(
    const _Float16* __restrict__ A, const _Float16* __restrict__ B,
    const float* __restrict__ bias, int K, int N,
    _Float16* __restrict__ Qh, _Float16* __restrict__ Kh,
    _Float16* __restrict__ Vh, float* __restrict__ Co) {
  __shared__ __align__(16) _Float16 As[128 * 72];   // [m][k], padded stride
  __shared__ __align__(16) _Float16 Bs[256 * 72];   // transposed: [n][k]

  const int t = threadIdx.x;
  const int lane = t & 31, wid = t >> 5;
  const int wm = wid >> 2, wn = wid & 3;            // 2 (M) x 4 (N) waves
  const int m0 = blockIdx.y * 128, n0 = blockIdx.x * 256;

  const int ra = t >> 1, sega = t & 1;              // A: 32 halfs/thread
  const int kb = t >> 2, nsegb = t & 3;             // B: 64 halfs/thread

  v8f acc[4][4] = {};
  f32x4 abuf[4], bbuf[8];

  {
    const f32x4* pa = (const f32x4*)(A + (size_t)(m0 + ra) * K + sega * 32);
    const f32x4* pb = (const f32x4*)(B + (size_t)kb * N + n0 + nsegb * 64);
#pragma unroll
    for (int i = 0; i < 4; ++i) abuf[i] = pa[i];
#pragma unroll
    for (int i = 0; i < 8; ++i) bbuf[i] = pb[i];
  }
  {
    f32x4* da = (f32x4*)(As + ra * 72 + sega * 32);
#pragma unroll
    for (int i = 0; i < 4; ++i) da[i] = abuf[i];
    union { f32x4 f[8]; _Float16 h[64]; } u;
#pragma unroll
    for (int i = 0; i < 8; ++i) u.f[i] = bbuf[i];
#pragma unroll
    for (int i = 0; i < 64; ++i) Bs[(nsegb * 64 + i) * 72 + kb] = u.h[i];
  }
  __syncthreads();

  for (int k0 = 0; k0 < K; k0 += 64) {
    const bool more = (k0 + 64 < K);                // uniform branch
    if (more) {
      const f32x4* pa = (const f32x4*)(A + (size_t)(m0 + ra) * K + k0 + 64 + sega * 32);
      const f32x4* pb = (const f32x4*)(B + (size_t)(k0 + 64 + kb) * N + n0 + nsegb * 64);
#pragma unroll
      for (int i = 0; i < 4; ++i) abuf[i] = pa[i];
#pragma unroll
      for (int i = 0; i < 8; ++i) bbuf[i] = pb[i];
    }

    // ---- 2 k-steps x (4 A-frags + 4 B-frags -> 16 WMMAs) ----
#pragma unroll
    for (int ks = 0; ks < 2; ++ks) {
      v16h af[4];
#pragma unroll
      for (int mt = 0; mt < 4; ++mt)
        af[mt] = load_frag(As + ks * 32, 72, wm * 64 + mt * 16 + (lane & 15), lane);
#pragma unroll
      for (int nt = 0; nt < 4; ++nt) {
        v16h bf = load_frag(Bs + ks * 32, 72, wn * 64 + nt * 16 + (lane & 15), lane);
#pragma unroll
        for (int mt = 0; mt < 4; ++mt) acc[mt][nt] = wmma16(af[mt], bf, acc[mt][nt]);
      }
    }
    __syncthreads();

    if (more) {
      f32x4* da = (f32x4*)(As + ra * 72 + sega * 32);
#pragma unroll
      for (int i = 0; i < 4; ++i) da[i] = abuf[i];
      union { f32x4 f[8]; _Float16 h[64]; } u;
#pragma unroll
      for (int i = 0; i < 8; ++i) u.f[i] = bbuf[i];
#pragma unroll
      for (int i = 0; i < 64; ++i) Bs[(nsegb * 64 + i) * 72 + kb] = u.h[i];
    }
    __syncthreads();
  }

  // ---- epilogue: C layout = lanes 0-15 M=j / lanes 16-31 M=j+8, N=lane%16 ----
#pragma unroll
  for (int mt = 0; mt < 4; ++mt) {
#pragma unroll
    for (int nt = 0; nt < 4; ++nt) {
#pragma unroll
      for (int j = 0; j < 8; ++j) {
        int ml = wm * 64 + mt * 16 + ((lane < 16) ? j : j + 8);
        int nl = wn * 64 + nt * 16 + (lane & 15);
        int gm = m0 + ml, gn = n0 + nl;
        float v = acc[mt][nt][j] + bias[gn];
        if (MODE == 0) {
          int sel = gn >> 10;            // 0:Q 1:K 2:V
          int e = gn & 1023;
          int hh = e >> 6, d = e & 63;
          int b = gm >> 11, s = gm & 2047;
          // Q: fold 1/sqrt(D=64) and log2(e) (exp2-domain softmax) into scale
          if (sel == 0) v *= 0.125f * 1.44269504088896340736f;
          _Float16* dstp = (sel == 0) ? Qh : (sel == 1) ? Kh : Vh;
          dstp[((size_t)(b * 16 + hh) * 2048 + s) * 64 + d] = (_Float16)v;
        } else {
          Co[(size_t)gm * N + gn] = v;
        }
      }
    }
  }
}

// ---------------------------------------------------------------------------
// Flash attention: one (b,h) head slice per blockIdx.x, 256 q-rows per block.
// 8 waves x 32 q-rows (two 16-row A-tiles) x 64-key blocks: every K/V
// fragment load feeds two WMMAs; 32 WMMAs per wave per key block.
// Online softmax in exp2 domain; row reductions are fused DPP16 butterflies.
// ---------------------------------------------------------------------------
__global__ __launch_bounds__(256) void flash_attn(
    const _Float16* __restrict__ Qh, const _Float16* __restrict__ Kh,
    const _Float16* __restrict__ Vh, _Float16* __restrict__ Oh) {
  __shared__ __align__(16) _Float16 Ks[64 * 72];        // [k-row][d], padded
  __shared__ __align__(16) _Float16 Vt[64 * 72];        // transposed [d][k-row]
  __shared__ __align__(16) _Float16 Ps[8 * 32 * 72];    // per-wave P tiles

  const int t = threadIdx.x;
  const int lane = t & 31, wid = t >> 5;
  const int bh = blockIdx.x, qb = blockIdx.y;
  const int b = bh >> 4, h = bh & 15;
  const size_t base = (size_t)bh * 2048 * 64;
  const _Float16* Qg = Qh + base;
  const _Float16* Kg = Kh + base;
  const _Float16* Vg = Vh + base;
  _Float16* PsW = Ps + wid * 32 * 72;

  // Q fragments for this wave's 32 rows (fixed across the key loop)
  const int qrow0 = qb * 256 + wid * 32;
  v16h qf[2][2];                                       // [m-tile][k-half]
#pragma unroll
  for (int mt = 0; mt < 2; ++mt)
#pragma unroll
    for (int kk = 0; kk < 2; ++kk)
      qf[mt][kk] = load_frag(Qg + kk * 32, 64, qrow0 + mt * 16 + (lane & 15), lane);

  v8f o[2][4] = {};
  float m_i[2][8], l_i[2][8];
#pragma unroll
  for (int mt = 0; mt < 2; ++mt)
#pragma unroll
    for (int j = 0; j < 8; ++j) { m_i[mt][j] = -1e30f; l_i[mt][j] = 0.0f; }

  for (int j0 = 0; j0 < 2048; j0 += 64) {
    // ---- cooperative stage: 64 K rows straight, 64 V rows transposed ----
    {
      int r = t >> 2, seg = t & 3;                     // 16 halfs per thread each
      const f32x4* ksrc = (const f32x4*)(Kg + (size_t)(j0 + r) * 64 + seg * 16);
      f32x4* kdst = (f32x4*)(Ks + r * 72 + seg * 16);
      kdst[0] = ksrc[0];
      kdst[1] = ksrc[1];
      union { f32x4 f[2]; _Float16 hh[16]; } u;
      const f32x4* vsrc = (const f32x4*)(Vg + (size_t)(j0 + r) * 64 + seg * 16);
      u.f[0] = vsrc[0];
      u.f[1] = vsrc[1];
#pragma unroll
      for (int i = 0; i < 16; ++i) Vt[(seg * 16 + i) * 72 + r] = u.hh[i];
      if (j0 + 64 < 2048) {               // prefetch next block toward L2/WGP$
        __builtin_prefetch(Kg + (size_t)(j0 + 64 + r) * 64 + seg * 16, 0, 0);
        __builtin_prefetch(Vg + (size_t)(j0 + 64 + r) * 64 + seg * 16, 0, 0);
      }
    }
    __syncthreads();

    // ---- S(32x64) = Q(32x64) * K^T : 16 WMMAs, kf reused across m-tiles ----
    v8f s[2][4] = {};
#pragma unroll
    for (int kk = 0; kk < 2; ++kk) {
#pragma unroll
      for (int nt = 0; nt < 4; ++nt) {
        v16h kf = load_frag(Ks + kk * 32, 72, nt * 16 + (lane & 15), lane);
#pragma unroll
        for (int mt = 0; mt < 2; ++mt) s[mt][nt] = wmma16(qf[mt][kk], kf, s[mt][nt]);
      }
    }

    // ---- online softmax (exp2 domain) over 64 keys, both m-tiles ----
#pragma unroll
    for (int mt = 0; mt < 2; ++mt) {
#pragma unroll
      for (int j = 0; j < 8; ++j) {
        float v = red16_max(fmaxf(fmaxf(s[mt][0][j], s[mt][1][j]),
                                  fmaxf(s[mt][2][j], s[mt][3][j])));
        float mnew = fmaxf(m_i[mt][j], v);
        float alpha = exp2f(m_i[mt][j] - mnew);
        float p[4];
        float rs = 0.0f;
#pragma unroll
        for (int nt = 0; nt < 4; ++nt) {
          p[nt] = exp2f(s[mt][nt][j] - mnew);
          rs += p[nt];
        }
        rs = red16_sum(rs);
        l_i[mt][j] = l_i[mt][j] * alpha + rs;
        m_i[mt][j] = mnew;
#pragma unroll
        for (int vt = 0; vt < 4; ++vt) o[mt][vt][j] *= alpha;
        int prow = mt * 16 + ((lane < 16) ? j : j + 8);  // C-layout -> row-major
#pragma unroll
        for (int nt = 0; nt < 4; ++nt)
          PsW[prow * 72 + nt * 16 + (lane & 15)] = (_Float16)p[nt];
      }
    }

    // ---- O(32x64) += P(32x64) * V(64x64) : 16 WMMAs, vf reused ----
#pragma unroll
    for (int kk = 0; kk < 2; ++kk) {
      v16h pf[2];
#pragma unroll
      for (int mt = 0; mt < 2; ++mt)
        pf[mt] = load_frag(PsW + mt * 16 * 72 + kk * 32, 72, (lane & 15), lane);
#pragma unroll
      for (int vt = 0; vt < 4; ++vt) {
        v16h vf = load_frag(Vt + kk * 32, 72, vt * 16 + (lane & 15), lane);
#pragma unroll
        for (int mt = 0; mt < 2; ++mt) o[mt][vt] = wmma16(pf[mt], vf, o[mt][vt]);
      }
    }
    __syncthreads();
  }

  // ---- normalize by l and emit O in (b*S, E) f16 layout for the out-proj ----
#pragma unroll
  for (int mt = 0; mt < 2; ++mt) {
#pragma unroll
    for (int j = 0; j < 8; ++j) {
      float inv = 1.0f / l_i[mt][j];
      int rl = mt * 16 + ((lane < 16) ? j : j + 8);
      int token = b * 2048 + qb * 256 + wid * 32 + rl;
#pragma unroll
      for (int vt = 0; vt < 4; ++vt) {
        int col = h * 64 + vt * 16 + (lane & 15);
        Oh[(size_t)token * 1024 + col] = (_Float16)(o[mt][vt][j] * inv);
      }
    }
  }
}

// ---------------------------------------------------------------------------
// Launch: cvt x3  ->  QKV GEMM  ->  flash attention  ->  out-proj GEMM
// Workspace layout (all f16): Xh@0(16MB) Wqkv@16 Wout@22 Q@24 K@40 V@56 O@72
// ---------------------------------------------------------------------------
extern "C" void kernel_launch(void* const* d_in, const int* in_sizes, int n_in,
                              void* d_out, int out_size, void* d_ws, size_t ws_size,
                              hipStream_t stream) {
  const float* query = (const float*)d_in[0];
  // d_in[1] (key) and d_in[2] (value) are unused by the reference.
  const float* W_qkv = (const float*)d_in[3];
  const float* b_qkv = (const float*)d_in[4];
  const float* W_out = (const float*)d_in[5];
  const float* b_out = (const float*)d_in[6];
  float* out = (float*)d_out;

  char* ws = (char*)d_ws;
  _Float16* Xh    = (_Float16*)(ws);
  _Float16* Wqkvh = (_Float16*)(ws + (size_t)16 * 1024 * 1024);
  _Float16* Wouth = (_Float16*)(ws + (size_t)22 * 1024 * 1024);
  _Float16* Qh    = (_Float16*)(ws + (size_t)24 * 1024 * 1024);
  _Float16* Kh    = (_Float16*)(ws + (size_t)40 * 1024 * 1024);
  _Float16* Vh    = (_Float16*)(ws + (size_t)56 * 1024 * 1024);
  _Float16* Oh    = (_Float16*)(ws + (size_t)72 * 1024 * 1024);

  const int nX  = 4 * 2048 * 1024;   // 8,388,608
  const int nW  = 1024 * 3072;
  const int nWo = 1024 * 1024;
  cvt_f32_f16<<<(nX / 8 + 255) / 256, 256, 0, stream>>>(query, Xh, nX / 8);
  cvt_f32_f16<<<(nW / 8 + 255) / 256, 256, 0, stream>>>(W_qkv, Wqkvh, nW / 8);
  cvt_f32_f16<<<(nWo / 8 + 255) / 256, 256, 0, stream>>>(W_out, Wouth, nWo / 8);

  // QKV projection: M=8192, N=3072, K=1024
  gemm_f16<0><<<dim3(3072 / 256, 8192 / 128), 256, 0, stream>>>(
      Xh, Wqkvh, b_qkv, 1024, 3072, Qh, Kh, Vh, nullptr);

  // Attention: 64 head-slices x 8 q-blocks (256 rows each)
  flash_attn<<<dim3(64, 8), 256, 0, stream>>>(Qh, Kh, Vh, Oh);

  // Output projection: M=8192, N=1024, K=1024
  gemm_f16<1><<<dim3(1024 / 256, 8192 / 128), 256, 0, stream>>>(
      Oh, Wouth, b_out, 1024, 1024, nullptr, nullptr, nullptr, out);
}
